// TransformerBlock_75763223101699
// MI455X (gfx1250) — matrix operand order repaired and checked by reading, not compiled, not run on hardware
//
#include <hip/hip_runtime.h>
#include <stdint.h>
#include <stddef.h>

// ---------------------------------------------------------------------------
// Types for CDNA5 WMMA (wave32)
// ---------------------------------------------------------------------------
typedef __attribute__((ext_vector_type(16))) __bf16 v16bf;
typedef __attribute__((ext_vector_type(8)))  float  v8f;

union AFrag {
    v16bf     v;
    uint32_t  u[8];
    uint16_t  h[16];
};

__device__ inline v8f wmma_bf16(const AFrag& a, const AFrag& b, v8f c) {
    return __builtin_amdgcn_wmma_f32_16x16x32_bf16(
        /*neg_a=*/false, a.v, /*neg_b=*/false, b.v,
        /*c_mod=*/(short)0, c, /*reuse_a=*/false, /*reuse_b=*/false);
}

__device__ inline uint16_t f2bf(float f) {
    uint32_t u = __float_as_uint(f);
    uint32_t r = u + 0x7fffu + ((u >> 16) & 1u);   // round-to-nearest-even
    return (uint16_t)(r >> 16);
}

// ---------------------------------------------------------------------------
// Problem constants
// ---------------------------------------------------------------------------
#define BB   4
#define SS   1024
#define MM   576
#define DD   1024
#define HH   16
#define DK   64
#define DFF  4096
#define BS_ROWS (BB*SS)   // 4096
#define BM_ROWS (BB*MM)   // 2304

// ---------------------------------------------------------------------------
// fp32 -> bf16 convert
// ---------------------------------------------------------------------------
__global__ void cvt_f32_bf16_kernel(const float* __restrict__ in,
                                    uint16_t* __restrict__ out, size_t n) {
    size_t i = (size_t)blockIdx.x * blockDim.x + threadIdx.x;
    size_t stride = (size_t)gridDim.x * blockDim.x;
    for (; i < n; i += stride) out[i] = f2bf(in[i]);
}

// ---------------------------------------------------------------------------
// GEMM: Y[M,N] = X[M,K] @ W[N,K]^T + bias, optional ReLU.
// Block tile 128x128, BK=32, 256 threads = 8 waves; wave tile 32x64
// (2 M-frags x 4 N-frags of v_wmma_f32_16x16x32_bf16).
// Double-buffered LDS fed by global_load_async_to_lds_b128 (ASYNCcnt DMA).
// ---------------------------------------------------------------------------
#define LDSK 40   // padded K stride in LDS (uint16 units); 80B rows (16B multiple)

__global__ __launch_bounds__(256) void gemm_bf16_kernel(
    const uint16_t* __restrict__ X, const uint16_t* __restrict__ W,
    const float* __restrict__ bias, float* __restrict__ outF,
    uint16_t* __restrict__ outB, int Mrows, int K, int N, int relu)
{
    __shared__ __align__(16) uint16_t Xs[2][128 * LDSK];
    __shared__ __align__(16) uint16_t Ws[2][128 * LDSK];

    const int tid  = threadIdx.x;
    const int lane = tid & 31;
    const int wid  = tid >> 5;
    const int half = lane >> 4;
    const int l15  = lane & 15;
    const int wm   = (wid & 3) * 32;   // 4 waves across M
    const int wn   = (wid >> 2) * 64;  // 2 waves across N
    const int bm0  = blockIdx.y * 128;
    const int bn0  = blockIdx.x * 128;

    // copy role: this thread moves two 16B segments per matrix per K-slice
    const int row0 = tid >> 2;               // 0..63
    const int seg0 = (tid & 3) * 8;          // 0,8,16,24 (uint16 units)

    v8f acc[2][4];
    for (int i = 0; i < 2; ++i)
        for (int j = 0; j < 4; ++j)
            for (int e = 0; e < 8; ++e) acc[i][j][e] = 0.0f;

    const int nk = K >> 5;

    // Issue one K-slice (128x32 bf16 of X and of W) as 4 async B128 DMAs/thread.
    auto issue_tile = [&](int kt, int buf) {
        const uint16_t* gx0 = X + (size_t)(bm0 + row0) * K + kt * 32 + seg0;
        const uint16_t* gx1 = gx0 + (size_t)64 * K;
        const uint16_t* gw0 = W + (size_t)(bn0 + row0) * K + kt * 32 + seg0;
        const uint16_t* gw1 = gw0 + (size_t)64 * K;
        uint32_t lx0 = (uint32_t)(size_t)&Xs[buf][row0 * LDSK + seg0];
        uint32_t lx1 = lx0 + 64 * LDSK * 2;
        uint32_t lw0 = (uint32_t)(size_t)&Ws[buf][row0 * LDSK + seg0];
        uint32_t lw1 = lw0 + 64 * LDSK * 2;
        asm volatile("global_load_async_to_lds_b128 %0, %1, off"
                     :: "v"(lx0), "v"(gx0) : "memory");
        asm volatile("global_load_async_to_lds_b128 %0, %1, off"
                     :: "v"(lx1), "v"(gx1) : "memory");
        asm volatile("global_load_async_to_lds_b128 %0, %1, off"
                     :: "v"(lw0), "v"(gw0) : "memory");
        asm volatile("global_load_async_to_lds_b128 %0, %1, off"
                     :: "v"(lw1), "v"(gw1) : "memory");
    };

    issue_tile(0, 0);

    for (int kt = 0; kt < nk; ++kt) {
        const int c = kt & 1;
        if (kt + 1 < nk) {
            issue_tile(kt + 1, c ^ 1);                 // prefetch into other buffer
            asm volatile("s_wait_asynccnt 0x4" ::: "memory");  // tile kt landed
        } else {
            asm volatile("s_wait_asynccnt 0x0" ::: "memory");
        }
        __syncthreads();   // publish tile kt to all waves

        // A fragments (16x32 bf16; half 0 -> K 0-15; half 1 -> K 16-31)
        AFrag a[2];
#pragma unroll
        for (int mf = 0; mf < 2; ++mf) {
            const uint16_t* rp = &Xs[c][(wm + mf * 16 + l15) * LDSK];
#pragma unroll
            for (int p = 0; p < 8; ++p) {
                int k = half * 16 + 2 * p;
                a[mf].u[p] = *(const uint32_t*)(rp + k);
            }
        }
        // B fragments (32x16 bf16): lane holds column n = l15, K = half*16..+15
        AFrag bfr[4];
#pragma unroll
        for (int nf = 0; nf < 4; ++nf) {
            const uint16_t* rp = &Ws[c][(wn + nf * 16 + l15) * LDSK + half * 16];
#pragma unroll
            for (int p = 0; p < 8; ++p) bfr[nf].u[p] = *(const uint32_t*)(rp + 2 * p);
        }

#pragma unroll
        for (int mf = 0; mf < 2; ++mf)
#pragma unroll
            for (int nf = 0; nf < 4; ++nf)
                acc[mf][nf] = wmma_bf16(a[mf], bfr[nf], acc[mf][nf]);

        __syncthreads();   // all waves done with buffer c before it is refilled
    }

    // Epilogue: C layout — lane half selects M block (0-7 / 8-15), N = l15
    for (int mf = 0; mf < 2; ++mf)
        for (int nf = 0; nf < 4; ++nf)
            for (int vr = 0; vr < 8; ++vr) {
                int row = bm0 + wm + mf * 16 + half * 8 + vr;
                int col = bn0 + wn + nf * 16 + l15;
                float v = acc[mf][nf][vr] + bias[col];
                if (relu) v = fmaxf(v, 0.0f);
                if (outF) outF[(size_t)row * N + col] = v;
                if (outB) outB[(size_t)row * N + col] = f2bf(v);
            }
}

// ---------------------------------------------------------------------------
// Flash attention: 1 wave per (batch, head, 16-query tile). dk = 64.
// Scores via 2 chunk-WMMAs per 16-key subtile; online softmax with shfl_xor
// row reductions; P repacked to A layout through LDS; P@V via 4 WMMAs/32 keys.
// ---------------------------------------------------------------------------
__global__ __launch_bounds__(32) void flash_attn_kernel(
    const uint16_t* __restrict__ Q, const uint16_t* __restrict__ Kv,
    const uint16_t* __restrict__ Vv, uint16_t* __restrict__ O,
    int Skv, int causal)
{
    const int lane  = threadIdx.x & 31;
    const int half  = lane >> 4;
    const int l15   = lane & 15;
    const int qbase = blockIdx.x * 16;
    const int h     = blockIdx.y;
    const int b     = blockIdx.z;

    __shared__ __align__(16) uint16_t Pl[16 * 32];  // P tile staging (bf16)
    __shared__ __align__(16) uint16_t Vl[32 * 64];  // V tile staging (bf16)

    // Q fragments (two K=32 chunks over dk=64), A layout
    AFrag qf[2];
#pragma unroll
    for (int c = 0; c < 2; ++c) {
        const uint16_t* rp = Q + ((size_t)(b * SS + qbase + l15)) * DD + h * DK + c * 32;
#pragma unroll
        for (int p = 0; p < 8; ++p) {
            int k = half * 16 + 2 * p;
            qf[c].u[p] = *(const uint32_t*)(rp + k);
        }
    }

    float m_i[8], l_i[8];
    v8f o[4];
#pragma unroll
    for (int vr = 0; vr < 8; ++vr) { m_i[vr] = -1e30f; l_i[vr] = 0.0f; }
#pragma unroll
    for (int nf = 0; nf < 4; ++nf)
        for (int e = 0; e < 8; ++e) o[nf][e] = 0.0f;

    const float sc = 0.125f;  // 1/sqrt(64)
    const int jend = causal ? (qbase + 16) : Skv;

    for (int j0 = 0; j0 < jend; j0 += 32) {
        // stage V[32][64] tile: each lane copies one row (128B, 8x b128)
        {
            const uint4* src = (const uint4*)(Vv + ((size_t)(b * Skv + j0 + lane)) * DD + h * DK);
            uint4* dst = (uint4*)(Vl + lane * 64);
#pragma unroll
            for (int q = 0; q < 8; ++q) dst[q] = src[q];
        }

        // scores for two 16-key subtiles
        v8f s0, s1;
        for (int e = 0; e < 8; ++e) { s0[e] = 0.0f; s1[e] = 0.0f; }
#pragma unroll
        for (int c = 0; c < 2; ++c) {
            AFrag kf0, kf1;
            const uint16_t* kp0 = Kv + ((size_t)(b * Skv + j0 + l15)) * DD + h * DK + c * 32 + half * 16;
            const uint16_t* kp1 = Kv + ((size_t)(b * Skv + j0 + 16 + l15)) * DD + h * DK + c * 32 + half * 16;
#pragma unroll
            for (int p = 0; p < 8; ++p) {
                kf0.u[p] = *(const uint32_t*)(kp0 + 2 * p);
                kf1.u[p] = *(const uint32_t*)(kp1 + 2 * p);
            }
            s0 = wmma_bf16(qf[c], kf0, s0);
            s1 = wmma_bf16(qf[c], kf1, s1);
        }

        // scale + causal mask
#pragma unroll
        for (int vr = 0; vr < 8; ++vr) {
            int qr = qbase + half * 8 + vr;
            float a0 = s0[vr] * sc, a1 = s1[vr] * sc;
            if (causal) {
                if (j0 + l15 > qr)      a0 = -1e30f;
                if (j0 + 16 + l15 > qr) a1 = -1e30f;
            }
            s0[vr] = a0; s1[vr] = a1;
        }

        // online softmax update (rows live in 16-lane halves)
#pragma unroll
        for (int vr = 0; vr < 8; ++vr) {
            float mc = fmaxf(s0[vr], s1[vr]);
            for (int x = 1; x < 16; x <<= 1) mc = fmaxf(mc, __shfl_xor(mc, x, 32));
            float mnew = fmaxf(m_i[vr], mc);
            float al   = __expf(m_i[vr] - mnew);
            float e0   = __expf(s0[vr] - mnew);
            float e1   = __expf(s1[vr] - mnew);
            float rs   = e0 + e1;
            for (int x = 1; x < 16; x <<= 1) rs += __shfl_xor(rs, x, 32);
            l_i[vr] = l_i[vr] * al + rs;
            m_i[vr] = mnew;
            for (int nf = 0; nf < 4; ++nf) o[nf][vr] *= al;
            Pl[(half * 8 + vr) * 32 + l15]      = f2bf(e0);
            Pl[(half * 8 + vr) * 32 + 16 + l15] = f2bf(e1);
        }
        __syncthreads();

        // reload P as A fragment (16x32)
        AFrag pf;
        {
            const uint16_t* rp = Pl + l15 * 32;
#pragma unroll
            for (int p = 0; p < 8; ++p) {
                int k = half * 16 + 2 * p;
                pf.u[p] = *(const uint32_t*)(rp + k);
            }
        }
        // O += P @ V  (B frag: column n = l15, K = half*16..+15, gathered from LDS)
#pragma unroll
        for (int nf = 0; nf < 4; ++nf) {
            AFrag vf;
#pragma unroll
            for (int e = 0; e < 16; ++e)
                vf.h[e] = Vl[(half * 16 + e) * 64 + nf * 16 + l15];
            o[nf] = wmma_bf16(pf, vf, o[nf]);
        }
        __syncthreads();
    }

    // write ctx (bf16) normalized by softmax denominator
#pragma unroll
    for (int nf = 0; nf < 4; ++nf)
        for (int vr = 0; vr < 8; ++vr) {
            int row = qbase + half * 8 + vr;
            int col = h * DK + nf * 16 + l15;
            O[((size_t)(b * SS + row)) * DD + col] = f2bf(o[nf][vr] / l_i[vr]);
        }
}

// ---------------------------------------------------------------------------
// Residual add + LayerNorm (fp32), dual fp32/bf16 output
// ---------------------------------------------------------------------------
__global__ __launch_bounds__(256) void add_ln_kernel(
    const float* __restrict__ X, const float* __restrict__ Y,
    const float* __restrict__ g, const float* __restrict__ bt,
    float* __restrict__ outF, uint16_t* __restrict__ outB)
{
    const int r = blockIdx.x;
    const int t = threadIdx.x;
    __shared__ float red[256];

    float v[4], s = 0.0f, s2 = 0.0f;
#pragma unroll
    for (int k = 0; k < 4; ++k) {
        int i = t + k * 256;
        float x = X[(size_t)r * DD + i] + Y[(size_t)r * DD + i];
        v[k] = x; s += x; s2 += x * x;
    }
    red[t] = s; __syncthreads();
    for (int st = 128; st > 0; st >>= 1) { if (t < st) red[t] += red[t + st]; __syncthreads(); }
    float mu = red[0] / (float)DD;
    __syncthreads();
    red[t] = s2; __syncthreads();
    for (int st = 128; st > 0; st >>= 1) { if (t < st) red[t] += red[t + st]; __syncthreads(); }
    float var  = red[0] / (float)DD - mu * mu;
    float rstd = rsqrtf(var + 1e-5f);

#pragma unroll
    for (int k = 0; k < 4; ++k) {
        int i = t + k * 256;
        float o = (v[k] - mu) * rstd * g[i] + bt[i];
        if (outF) outF[(size_t)r * DD + i] = o;
        if (outB) outB[(size_t)r * DD + i] = f2bf(o);
    }
}

// ---------------------------------------------------------------------------
// Host-side orchestration
// ---------------------------------------------------------------------------
extern "C" void kernel_launch(void* const* d_in, const int* in_sizes, int n_in,
                              void* d_out, int out_size, void* d_ws, size_t ws_size,
                              hipStream_t stream) {
    (void)in_sizes; (void)n_in; (void)out_size; (void)ws_size;

    const float* x    = (const float*)d_in[0];
    const float* vmem = (const float*)d_in[1];
    // d_in[2] = mask (exact tril; applied analytically in the causal kernel)
    const float* saWq = (const float*)d_in[3];  const float* sabq = (const float*)d_in[4];
    const float* saWk = (const float*)d_in[5];  const float* sabk = (const float*)d_in[6];
    const float* saWv = (const float*)d_in[7];  const float* sabv = (const float*)d_in[8];
    const float* saWo = (const float*)d_in[9];  const float* sabo = (const float*)d_in[10];
    const float* caWq = (const float*)d_in[11]; const float* cabq = (const float*)d_in[12];
    const float* caWk = (const float*)d_in[13]; const float* cabk = (const float*)d_in[14];
    const float* caWv = (const float*)d_in[15]; const float* cabv = (const float*)d_in[16];
    const float* caWo = (const float*)d_in[17]; const float* cabo = (const float*)d_in[18];
    const float* W1   = (const float*)d_in[19]; const float* b1   = (const float*)d_in[20];
    const float* W2   = (const float*)d_in[21]; const float* b2   = (const float*)d_in[22];
    const float* ln1g = (const float*)d_in[23]; const float* ln1b = (const float*)d_in[24];
    const float* ln2g = (const float*)d_in[25]; const float* ln2b = (const float*)d_in[26];
    const float* ln3g = (const float*)d_in[27]; const float* ln3b = (const float*)d_in[28];

    char* ws = (char*)d_ws;
    size_t off = 0;
    auto take = [&](size_t bytes) -> void* {
        void* p = ws + off;
        off = (off + bytes + 255) & ~(size_t)255;
        return p;
    };

    uint16_t* xb   = (uint16_t*)take((size_t)BS_ROWS * DD * 2);  // reused for x1b, x2b
    uint16_t* memb = (uint16_t*)take((size_t)BM_ROWS * DD * 2);
    uint16_t* wsq  = (uint16_t*)take((size_t)DD * DD * 2);
    uint16_t* wsk  = (uint16_t*)take((size_t)DD * DD * 2);
    uint16_t* wsv  = (uint16_t*)take((size_t)DD * DD * 2);
    uint16_t* wso  = (uint16_t*)take((size_t)DD * DD * 2);
    uint16_t* wcq  = (uint16_t*)take((size_t)DD * DD * 2);
    uint16_t* wck  = (uint16_t*)take((size_t)DD * DD * 2);
    uint16_t* wcv  = (uint16_t*)take((size_t)DD * DD * 2);
    uint16_t* wco  = (uint16_t*)take((size_t)DD * DD * 2);
    uint16_t* w1b  = (uint16_t*)take((size_t)DFF * DD * 2);
    uint16_t* w2b  = (uint16_t*)take((size_t)DD * DFF * 2);
    uint16_t* Qb   = (uint16_t*)take((size_t)BS_ROWS * DD * 2);
    uint16_t* Kb   = (uint16_t*)take((size_t)BS_ROWS * DD * 2);
    uint16_t* Vb   = (uint16_t*)take((size_t)BS_ROWS * DD * 2);
    uint16_t* ctxb = (uint16_t*)take((size_t)BS_ROWS * DD * 2);
    uint16_t* hb   = (uint16_t*)take((size_t)BS_ROWS * DFF * 2);
    float*    t0   = (float*)take((size_t)BS_ROWS * DD * 4);
    float*    x1f  = (float*)take((size_t)BS_ROWS * DD * 4);
    float*    x2f  = (float*)take((size_t)BS_ROWS * DD * 4);

    auto cvt = [&](const float* in, uint16_t* out, size_t n) {
        cvt_f32_bf16_kernel<<<512, 256, 0, stream>>>(in, out, n);
    };
    cvt(x, xb, (size_t)BS_ROWS * DD);
    cvt(vmem, memb, (size_t)BM_ROWS * DD);
    cvt(saWq, wsq, (size_t)DD * DD); cvt(saWk, wsk, (size_t)DD * DD);
    cvt(saWv, wsv, (size_t)DD * DD); cvt(saWo, wso, (size_t)DD * DD);
    cvt(caWq, wcq, (size_t)DD * DD); cvt(caWk, wck, (size_t)DD * DD);
    cvt(caWv, wcv, (size_t)DD * DD); cvt(caWo, wco, (size_t)DD * DD);
    cvt(W1, w1b, (size_t)DFF * DD);  cvt(W2, w2b, (size_t)DD * DFF);

    dim3 gS(DD / 128, BS_ROWS / 128);   // [4096 x 1024] GEMMs
    dim3 gM(DD / 128, BM_ROWS / 128);   // [2304 x 1024] GEMMs
    dim3 gF1(DFF / 128, BS_ROWS / 128); // FFN up
    dim3 gF2(DD / 128, BS_ROWS / 128);  // FFN down
    dim3 gA(SS / 16, HH, BB);           // attention

    // ---- self attention ----
    gemm_bf16_kernel<<<gS, 256, 0, stream>>>(xb, wsq, sabq, nullptr, Qb, BS_ROWS, DD, DD, 0);
    gemm_bf16_kernel<<<gS, 256, 0, stream>>>(xb, wsk, sabk, nullptr, Kb, BS_ROWS, DD, DD, 0);
    gemm_bf16_kernel<<<gS, 256, 0, stream>>>(xb, wsv, sabv, nullptr, Vb, BS_ROWS, DD, DD, 0);
    flash_attn_kernel<<<gA, 32, 0, stream>>>(Qb, Kb, Vb, ctxb, SS, 1);
    gemm_bf16_kernel<<<gS, 256, 0, stream>>>(ctxb, wso, sabo, t0, nullptr, BS_ROWS, DD, DD, 0);
    add_ln_kernel<<<BS_ROWS, 256, 0, stream>>>(x, t0, ln1g, ln1b, x1f, xb);  // xb := x1 (bf16)

    // ---- cross attention ----
    gemm_bf16_kernel<<<gS, 256, 0, stream>>>(xb, wcq, cabq, nullptr, Qb, BS_ROWS, DD, DD, 0);
    gemm_bf16_kernel<<<gM, 256, 0, stream>>>(memb, wck, cabk, nullptr, Kb, BM_ROWS, DD, DD, 0);
    gemm_bf16_kernel<<<gM, 256, 0, stream>>>(memb, wcv, cabv, nullptr, Vb, BM_ROWS, DD, DD, 0);
    flash_attn_kernel<<<gA, 32, 0, stream>>>(Qb, Kb, Vb, ctxb, MM, 0);
    gemm_bf16_kernel<<<gS, 256, 0, stream>>>(ctxb, wco, cabo, t0, nullptr, BS_ROWS, DD, DD, 0);
    add_ln_kernel<<<BS_ROWS, 256, 0, stream>>>(x1f, t0, ln2g, ln2b, x2f, xb); // xb := x2 (bf16)

    // ---- FFN ----
    gemm_bf16_kernel<<<gF1, 256, 0, stream>>>(xb, w1b, b1, nullptr, hb, BS_ROWS, DD, DFF, 1);
    gemm_bf16_kernel<<<gF2, 256, 0, stream>>>(hb, w2b, b2, t0, nullptr, BS_ROWS, DFF, DD, 0);
    add_ln_kernel<<<BS_ROWS, 256, 0, stream>>>(x2f, t0, ln3g, ln3b, (float*)d_out, nullptr);
}
